// MHERNN_21715354648616
// MI455X (gfx1250) — matrix operand order repaired and checked by reading, not compiled, not run on hardware
//
#include <hip/hip_runtime.h>
#include <hip/hip_bf16.h>
#include <stdint.h>

// Problem constants (from reference)
#define INPUT_SIZE 64
#define HIDDEN     256
#define GATES      768            // 3*HIDDEN
#define KTOT       320            // INPUT_SIZE + HIDDEN (fused A = [x_t | h])
#define BATCH      512
#define SEQ        1024
#define NCLS       10
#define BTILE      16             // batch rows per workgroup (= WMMA M)
#define A_STRIDE   328            // padded LDS row stride (bf16 elems): 656B, 16B-aligned, bank-spread

typedef __bf16 bf16_t;
typedef bf16_t v16bf __attribute__((ext_vector_type(16)));
typedef bf16_t v8bf  __attribute__((ext_vector_type(8)));
typedef float  v8f   __attribute__((ext_vector_type(8)));

static __device__ __forceinline__ v8f wmma_bf16(v16bf a, v16bf b, v8f c) {
  // (neg_a, A, neg_b, B, c_mod, C, reuse_a, reuse_b)
  return __builtin_amdgcn_wmma_f32_16x16x32_bf16(false, a, false, b, (short)0, c,
                                                 false, false);
}

static __device__ __forceinline__ v16bf cat16(v8bf lo, v8bf hi) {
  return __builtin_shufflevector(lo, hi, 0,1,2,3,4,5,6,7,8,9,10,11,12,13,14,15);
}

// ---------------------------------------------------------------------------
// Prep: fuse Wx [768,64] and Wh [768,256] into bf16 Wcat [768][320]
// so gate = [x_t | h] @ Wcat[g]^T.  Lives in d_ws (480 KB), L2-resident.
// ---------------------------------------------------------------------------
__global__ void gru_prep_wcat(const float* __restrict__ Wx,
                              const float* __restrict__ Wh,
                              bf16_t* __restrict__ Wcat) {
  int idx = blockIdx.x * blockDim.x + threadIdx.x;
  if (idx >= GATES * KTOT) return;
  int g = idx / KTOT;
  int c = idx - g * KTOT;
  float v = (c < INPUT_SIZE) ? Wx[g * INPUT_SIZE + c]
                             : Wh[g * HIDDEN + (c - INPUT_SIZE)];
  Wcat[idx] = (bf16_t)v;
}

// ---------------------------------------------------------------------------
// Main: each workgroup owns 16 batch rows and runs the full 1024-step scan.
// 512 threads = 16 wave32s; wave w owns hidden columns [16w, 16w+16) for all
// three gates. Per step per wave: 30 x v_wmma_f32_16x16x32_bf16.
// x_{s+1} is async-copied to LDS during step s's compute (2 barriers/step).
// ---------------------------------------------------------------------------
__global__ __launch_bounds__(512)
void gru_scan(const float* __restrict__ x,        // [512,1024,64] f32
              const bf16_t* __restrict__ Wcat,    // [768,320] bf16
              const float* __restrict__ bias,     // [768]
              const float* __restrict__ fc_w,     // [10,256]
              const float* __restrict__ fc_b,     // [10]
              float* __restrict__ out) {          // [512,10]
  __shared__ __align__(16) bf16_t Alds[BTILE][A_STRIDE]; // cols 0..63 = x_t, 64..319 = h
  __shared__ __align__(16) float  Xs[BTILE][INPUT_SIZE]; // async-load staging (f32)

  const int tid   = threadIdx.x;
  const int wave  = tid >> 5;
  const int lane  = tid & 31;
  const int m     = lane & 15;        // A row / B column slot
  const int kh    = lane >> 4;        // K-half selector in fragment layouts
  const int bbase = blockIdx.x * BTILE;

  // h0 = 0
  for (int i = tid; i < BTILE * HIDDEN; i += 512)
    Alds[i >> 8][INPUT_SIZE + (i & 255)] = (bf16_t)0.f;

  // Per-lane gate column j and its biases (constant over time)
  const int j   = (wave << 4) + m;
  const float br = bias[j];
  const float bz = bias[HIDDEN + j];
  const float bn = bias[2 * HIDDEN + j];

  // Per-wave/lane B base pointers (row = gate base + column slot m,
  // K-half offset folded in once; per-k-step offsets are immediates)
  const bf16_t* Wr = Wcat + (size_t)((wave << 4) + m) * KTOT + kh * 16;
  const bf16_t* Wz = Wr + (size_t)HIDDEN * KTOT;
  const bf16_t* Wn = Wz + (size_t)HIDDEN * KTOT;

  // Async x-tile issue: threads 0..255 each copy 16B of the 16x64-f32 tile
  const int xr  = tid >> 4;            // staging row
  const int xc4 = (tid & 15) << 2;     // 4 floats
  const float* xrow = x + (size_t)(bbase + xr) * SEQ * INPUT_SIZE + xc4;
  const uint32_t xstage_lp = (uint32_t)(uintptr_t)&Xs[xr][xc4];

  // ---- preload x_0 into staging, convert into A panel ----
  if (tid < 256) {
    asm volatile("global_load_async_to_lds_b128 %0, %1, off"
                 :: "v"(xstage_lp), "v"(xrow) : "memory");
  }
  asm volatile("s_wait_asynccnt 0" ::: "memory");
  __syncthreads();
  {
    const float2 v2 = *(const float2*)&Xs[wave][lane << 1];
    Alds[wave][(lane << 1) + 0] = (bf16_t)v2.x;
    Alds[wave][(lane << 1) + 1] = (bf16_t)v2.y;
  }

  for (int s = 0; s < SEQ; ++s) {
    __syncthreads();   // A panel ready: x_s (cols 0..63) + h_s (cols 64..319)

    // -- issue async copy of x_{s+1} (clamped) now; waited only after WMMAs
    {
      const int sn = (s + 1 < SEQ) ? s + 1 : s;
      if (tid < 256) {
        const float* gp = xrow + (size_t)sn * INPUT_SIZE;
        asm volatile("global_load_async_to_lds_b128 %0, %1, off"
                     :: "v"(xstage_lp), "v"(gp) : "memory");
      }
    }

    // -- gate GEMMs, branch-free accumulator split:
    //    kb 0..1  : A cols [0,64)   (x_t)  -> acc_r, acc_z, acc_in
    //    kb 2..9  : A cols [64,320) (h)    -> acc_r, acc_z, acc_hn
    v8f acc_r = {}, acc_z = {}, acc_in = {}, acc_hn = {};
#pragma unroll
    for (int kb = 0; kb < 2; ++kb) {
      const int c0 = kb << 5;
      v16bf a = cat16(*(const v8bf*)&Alds[m][c0 + kh * 16],
                      *(const v8bf*)&Alds[m][c0 + 8 + kh * 16]);
      v16bf b_r = cat16(*(const v8bf*)(Wr + c0), *(const v8bf*)(Wr + c0 + 8));
      v16bf b_z = cat16(*(const v8bf*)(Wz + c0), *(const v8bf*)(Wz + c0 + 8));
      v16bf b_n = cat16(*(const v8bf*)(Wn + c0), *(const v8bf*)(Wn + c0 + 8));
      acc_r  = wmma_bf16(a, b_r, acc_r);
      acc_z  = wmma_bf16(a, b_z, acc_z);
      acc_in = wmma_bf16(a, b_n, acc_in);
    }
#pragma unroll 2
    for (int kb = 2; kb < KTOT / 32; ++kb) {
      const int c0 = kb << 5;
      v16bf a = cat16(*(const v8bf*)&Alds[m][c0 + kh * 16],
                      *(const v8bf*)&Alds[m][c0 + 8 + kh * 16]);
      v16bf b_r = cat16(*(const v8bf*)(Wr + c0), *(const v8bf*)(Wr + c0 + 8));
      v16bf b_z = cat16(*(const v8bf*)(Wz + c0), *(const v8bf*)(Wz + c0 + 8));
      v16bf b_n = cat16(*(const v8bf*)(Wn + c0), *(const v8bf*)(Wn + c0 + 8));
      acc_r  = wmma_bf16(a, b_r, acc_r);
      acc_z  = wmma_bf16(a, b_z, acc_z);
      acc_hn = wmma_bf16(a, b_n, acc_hn);
    }

    // -- pointwise GRU update on the D-tile (VGPR v -> row v + 8*kh, col j)
    float hnew[8];
#pragma unroll
    for (int v = 0; v < 8; ++v) {
      const int   mr   = v + (kh << 3);
      const float hold = (float)Alds[mr][INPUT_SIZE + j];
      const float rg = 1.f / (1.f + __expf(-(acc_r[v] + br)));
      const float zg = 1.f / (1.f + __expf(-(acc_z[v] + bz)));
      const float pre = acc_in[v] + rg * acc_hn[v] + bn;
      const float ng  = 1.f - 2.f / (1.f + __expf(2.f * pre));  // tanh(pre)
      hnew[v] = ng + zg * (hold - ng);
    }

    // Own async op (x_{s+1} -> staging) must be complete before the barrier
    // publishes the staging buffer to all waves.
    asm volatile("s_wait_asynccnt 0" ::: "memory");
    __syncthreads();   // all reads of A done this step + staging ready

    // -- single write phase: h' into cols [64,320), x_{s+1} into cols [0,64)
#pragma unroll
    for (int v = 0; v < 8; ++v)
      Alds[v + (kh << 3)][INPUT_SIZE + j] = (bf16_t)hnew[v];
    {
      const float2 v2 = *(const float2*)&Xs[wave][lane << 1];
      Alds[wave][(lane << 1) + 0] = (bf16_t)v2.x;
      Alds[wave][(lane << 1) + 1] = (bf16_t)v2.y;
    }
  }

  __syncthreads();

  // Final FC: [16,256] x [256,10] — tiny, scalar per-output
  if (tid < BTILE * NCLS) {
    const int r = tid / NCLS;
    const int c = tid - r * NCLS;
    float acc = fc_b[c];
    for (int k = 0; k < HIDDEN; ++k)
      acc += (float)Alds[r][INPUT_SIZE + k] * fc_w[c * HIDDEN + k];
    out[(size_t)(bbase + r) * NCLS + c] = acc;
  }
}

// ---------------------------------------------------------------------------
extern "C" void kernel_launch(void* const* d_in, const int* in_sizes, int n_in,
                              void* d_out, int out_size, void* d_ws, size_t ws_size,
                              hipStream_t stream) {
  const float* x    = (const float*)d_in[0];
  const float* Wx   = (const float*)d_in[1];
  const float* Wh   = (const float*)d_in[2];
  const float* bias = (const float*)d_in[3];
  const float* fc_w = (const float*)d_in[4];
  const float* fc_b = (const float*)d_in[5];
  float* out = (float*)d_out;

  bf16_t* Wcat = (bf16_t*)d_ws;                       // 768*320*2 = 480 KB

  const int total = GATES * KTOT;
  gru_prep_wcat<<<(total + 255) / 256, 256, 0, stream>>>(Wx, Wh, Wcat);
  gru_scan<<<BATCH / BTILE, 512, 0, stream>>>(x, Wcat, bias, fc_w, fc_b, out);
}
